// CGCNN_83751862272706
// MI455X (gfx1250) — compile-verified, matrix operand order checked by reading, not run on hardware
//
#include <hip/hip_runtime.h>
#include <hip/hip_bf16.h>

typedef __attribute__((ext_vector_type(16))) _Float16 v16h;
typedef __attribute__((ext_vector_type(8)))  float    v8f;

#define N_NODES  50000
#define N_EDGES  600000
#define N_GRAPHS 128
#define CH       128
#define KDIM     320
#define NC       (N_NODES * CH)
#define KS       328            // padded K stride (halfs): 656B rows, 16B aligned
#define EDGE_BLOCKS (N_EDGES / 64)   // 9375, exact

// ---------------------------------------------------------------- helpers
__device__ __forceinline__ void atomic_add_f32(float* p, float v) {
#if defined(__HIP_DEVICE_COMPILE__)
    unsafeAtomicAdd(p, v);
#else
    (void)p; (void)v;
#endif
}

// Gather one A fragment (16 edges x 32 K of z = [h[dst] | h[src] | edge_attr])
__device__ __forceinline__ v16h load_a_frag(const float* __restrict__ h,
                                            const float* __restrict__ ea,
                                            const int* sdst, const int* ssrc,
                                            int mbase, long long ebase,
                                            int lane, int kt)
{
    const int r  = lane & 15;
    const int kg = lane >> 4;
    union { _Float16 e[16]; v16h v; } A;
    #pragma unroll
    for (int c = 0; c < 2; ++c) {
        const int k0 = kt * 32 + c * 16 + kg * 8;   // chunk of 8 consecutive K
        const float* p;
        if (k0 < 128)       p = h  + (size_t)sdst[mbase + r] * CH + k0;
        else if (k0 < 256)  p = h  + (size_t)ssrc[mbase + r] * CH + (k0 - 128);
        else                p = ea + (size_t)(ebase + r) * 64   + (k0 - 256);
        const float4 u0 = reinterpret_cast<const float4*>(p)[0];
        const float4 u1 = reinterpret_cast<const float4*>(p)[1];
        const int o = c * 8;
        A.e[o+0] = (_Float16)u0.x; A.e[o+1] = (_Float16)u0.y;
        A.e[o+2] = (_Float16)u0.z; A.e[o+3] = (_Float16)u0.w;
        A.e[o+4] = (_Float16)u1.x; A.e[o+5] = (_Float16)u1.y;
        A.e[o+6] = (_Float16)u1.z; A.e[o+7] = (_Float16)u1.w;
    }
    return A.v;
}

// B fragment from LDS-resident transposed f16 weights: column n, K-chunk of 32
__device__ __forceinline__ v16h load_b_frag(const _Float16* wT, int n, int kt, int kg)
{
    union { uint4 q[2]; v16h v; } B;
    const _Float16* p = wT + (size_t)n * KS + kt * 32 + kg * 8;
    B.q[0] = *reinterpret_cast<const uint4*>(p);        // K: base .. base+7
    B.q[1] = *reinterpret_cast<const uint4*>(p + 16);   // K: base+16 .. +23
    return B.v;
}

// ---------------------------------------------------------------- kernels
__global__ void cg_init_kernel(float* red6, float* pool, float* cnt)
{
    const int i = blockIdx.x * blockDim.x + threadIdx.x;
    if (i < 8)                   red6[i] = 0.f;
    if (i < N_GRAPHS * CH)       pool[i] = 0.f;
    if (i < N_GRAPHS)            cnt[i]  = 0.f;
}

__global__ void cg_fc1_kernel(const float* __restrict__ x,
                              const float* __restrict__ W1,
                              const float* __restrict__ b1,
                              float* __restrict__ h)
{
    const int stride = gridDim.x * blockDim.x;
    for (int i = blockIdx.x * blockDim.x + threadIdx.x; i < NC; i += stride) {
        const int n = i >> 7, c = i & 127;
        const float* xr = x + (size_t)n * 92;
        float acc = b1[c];
        #pragma unroll 4
        for (int k = 0; k < 92; ++k) acc = fmaf(xr[k], W1[k * CH + c], acc);
        h[i] = acc;
    }
}

__global__ __launch_bounds__(256)
void cg_edge_mlp_kernel(const float* __restrict__ h, float* __restrict__ hn,
                        const int* __restrict__ src, const int* __restrict__ dst,
                        const float* __restrict__ ea,
                        const float* __restrict__ Wf, const float* __restrict__ bf,
                        const float* __restrict__ Ws, const float* __restrict__ bs)
{
    extern __shared__ char smem[];
    _Float16* wTf = (_Float16*)smem;              // [128][KS] f16 (W_f transposed)
    _Float16* wTs = wTf + 128 * KS;               // [128][KS] f16 (W_s transposed)
    int* sdst = (int*)(wTs + 128 * KS);           // 64 dst ids of current block
    int* ssrc = sdst + 64;                        // 64 src ids

    const int tid = threadIdx.x;

    // Stage transposed f16 weights once per workgroup
    for (int i = tid; i < KDIM * CH; i += 256) {
        const int k = i >> 7, n = i & 127;        // W is [320][128] row-major
        wTf[(size_t)n * KS + k] = (_Float16)Wf[i];
        wTs[(size_t)n * KS + k] = (_Float16)Ws[i];
    }
    __syncthreads();

    const int lane  = tid & 31;
    const int w     = tid >> 5;
    const int mtile = w & 3;              // 16-edge tile within 64-edge block
    const int nbase = (w >> 2) * 64;      // column half: 0 or 64
    const int kg    = lane >> 4;
    const int nlane = lane & 15;

    float biasF[4], biasS[4];
    #pragma unroll
    for (int nt = 0; nt < 4; ++nt) {
        const int n = nbase + nt * 16 + nlane;
        biasF[nt] = bf[n];
        biasS[nt] = bs[n];
    }

    for (int blk = blockIdx.x; blk < EDGE_BLOCKS; blk += gridDim.x) {
        const long long eb = (long long)blk * 64;
        if (tid < 64) { sdst[tid] = dst[eb + tid]; ssrc[tid] = src[eb + tid]; }
        __syncthreads();

        v8f accF[4] = {}; v8f accS[4] = {};

        for (int kt = 0; kt < KDIM / 32; ++kt) {
            const v16h a = load_a_frag(h, ea, sdst, ssrc, mtile * 16,
                                       eb + mtile * 16, lane, kt);
            #pragma unroll
            for (int nt = 0; nt < 4; ++nt) {
                const int n = nbase + nt * 16 + nlane;
                const v16h bF = load_b_frag(wTf, n, kt, kg);
                accF[nt] = __builtin_amdgcn_wmma_f32_16x16x32_f16(
                    false, a, false, bF, (short)0, accF[nt], false, false);
                const v16h bS = load_b_frag(wTs, n, kt, kg);
                accS[nt] = __builtin_amdgcn_wmma_f32_16x16x32_f16(
                    false, a, false, bS, (short)0, accS[nt], false, false);
            }
        }

        // m = sigmoid(f) * softplus(s); scatter-add into hn[dst]
        #pragma unroll
        for (int nt = 0; nt < 4; ++nt) {
            const int n = nbase + nt * 16 + nlane;
            #pragma unroll
            for (int g = 0; g < 8; ++g) {
                const int m = g + kg * 8;             // D layout: VGPR g -> row
                const float f = accF[nt][g] + biasF[nt];
                const float s = accS[nt][g] + biasS[nt];
                const float sig = 1.f / (1.f + __expf(-f));
                const float sp  = fmaxf(s, 0.f) + __logf(1.f + __expf(-fabsf(s)));
                const int   d   = sdst[mtile * 16 + m];
                atomic_add_f32(&hn[(size_t)d * CH + n], sig * sp);
            }
        }
        __syncthreads();
    }
}

__global__ void cg_reduce_kernel(const float* __restrict__ v, float* __restrict__ red2)
{
    __shared__ float s1[256], s2[256];
    float a = 0.f, b = 0.f;
    const int stride = gridDim.x * 256;
    for (int i = blockIdx.x * 256 + threadIdx.x; i < NC; i += stride) {
        const float x = v[i];
        a += x; b += x * x;
    }
    s1[threadIdx.x] = a; s2[threadIdx.x] = b;
    __syncthreads();
    for (int o = 128; o > 0; o >>= 1) {
        if (threadIdx.x < o) {
            s1[threadIdx.x] += s1[threadIdx.x + o];
            s2[threadIdx.x] += s2[threadIdx.x + o];
        }
        __syncthreads();
    }
    if (threadIdx.x == 0) {
        atomic_add_f32(&red2[0], s1[0]);
        atomic_add_f32(&red2[1], s2[0]);
    }
}

__global__ void cg_norm_relu_kernel(const float* __restrict__ xin,
                                    const float* __restrict__ gw,
                                    const float* __restrict__ gb,
                                    const float* __restrict__ red2,
                                    float* __restrict__ out)
{
    const float inv  = 1.f / (float)NC;
    const float mean = red2[0] * inv;
    const float var  = red2[1] * inv - mean * mean;
    const float rs   = 1.f / (sqrtf(fmaxf(var, 0.f)) + 1e-5f);
    const int stride = gridDim.x * blockDim.x;
    for (int i = blockIdx.x * blockDim.x + threadIdx.x; i < NC; i += stride) {
        const int c = i & 127;
        const float v = (xin[i] - mean) * rs * gw[c] + gb[c];
        out[i] = fmaxf(v, 0.f);
    }
}

__global__ void cg_pool_kernel(const float* __restrict__ h,
                               const int* __restrict__ batch,
                               float* __restrict__ pool, float* __restrict__ cnt)
{
    const int stride = gridDim.x * blockDim.x;
    for (int i = blockIdx.x * blockDim.x + threadIdx.x; i < NC; i += stride) {
        const int n = i >> 7, c = i & 127;
        const int b = batch[n];
        atomic_add_f32(&pool[b * CH + c], h[i]);
        if (c == 0) atomic_add_f32(&cnt[b], 1.f);
    }
}

__global__ void cg_head_kernel(const float* __restrict__ pool,
                               const float* __restrict__ cnt,
                               const float* __restrict__ W2, const float* __restrict__ b2,
                               const float* __restrict__ W3, const float* __restrict__ b3,
                               float* __restrict__ out)
{
    const int g = threadIdx.x;
    if (g >= N_GRAPHS) return;
    const float invc = 1.f / fmaxf(cnt[g], 1.f);
    float o = b3[0];
    #pragma unroll
    for (int j = 0; j < 16; ++j) {
        float s = 0.f;
        for (int i = 0; i < CH; ++i)
            s = fmaf(pool[g * CH + i], W2[i * 16 + j], s);
        const float h2 = fmaxf(fmaf(s, invc, b2[j]), 0.f);
        o = fmaf(h2, W3[j], o);
    }
    out[g] = o;
}

// ---------------------------------------------------------------- launcher
extern "C" void kernel_launch(void* const* d_in, const int* in_sizes, int n_in,
                              void* d_out, int out_size, void* d_ws, size_t ws_size,
                              hipStream_t stream)
{
    (void)in_sizes; (void)n_in; (void)out_size; (void)ws_size;

    const float* x     = (const float*)d_in[0];
    const int*   ei    = (const int*)d_in[1];
    const float* ea    = (const float*)d_in[2];
    const int*   batch = (const int*)d_in[3];
    const float* W1    = (const float*)d_in[4];
    const float* b1    = (const float*)d_in[5];
    const float* Wf[3] = {(const float*)d_in[6],  (const float*)d_in[12], (const float*)d_in[18]};
    const float* bf[3] = {(const float*)d_in[7],  (const float*)d_in[13], (const float*)d_in[19]};
    const float* Ws[3] = {(const float*)d_in[8],  (const float*)d_in[14], (const float*)d_in[20]};
    const float* bs[3] = {(const float*)d_in[9],  (const float*)d_in[15], (const float*)d_in[21]};
    const float* gw[3] = {(const float*)d_in[10], (const float*)d_in[16], (const float*)d_in[22]};
    const float* gb[3] = {(const float*)d_in[11], (const float*)d_in[17], (const float*)d_in[23]};
    const float* W2 = (const float*)d_in[24];
    const float* b2 = (const float*)d_in[25];
    const float* W3 = (const float*)d_in[26];
    const float* b3 = (const float*)d_in[27];
    float* out = (float*)d_out;

    const int* src = ei;
    const int* dst = ei + N_EDGES;

    // workspace layout (floats)
    float* h    = (float*)d_ws;
    float* hn   = h  + NC;
    float* red  = hn + NC;                 // 8 floats (2 per layer, padded)
    float* pool = red + 64;
    float* cnt  = pool + N_GRAPHS * CH;

    const size_t smem = (size_t)2 * 128 * KS * 2 /*f16*/ + 2 * 64 * sizeof(int);

    cg_init_kernel<<<64, 256, 0, stream>>>(red, pool, cnt);
    cg_fc1_kernel<<<2048, 256, 0, stream>>>(x, W1, b1, h);

    for (int l = 0; l < 3; ++l) {
        hipMemcpyAsync(hn, h, (size_t)NC * sizeof(float),
                       hipMemcpyDeviceToDevice, stream);
        cg_edge_mlp_kernel<<<512, 256, smem, stream>>>(h, hn, src, dst, ea,
                                                       Wf[l], bf[l], Ws[l], bs[l]);
        cg_reduce_kernel<<<1024, 256, 0, stream>>>(hn, red + 2 * l);
        cg_norm_relu_kernel<<<2048, 256, 0, stream>>>(hn, gw[l], gb[l],
                                                      red + 2 * l, h);
    }

    cg_pool_kernel<<<2048, 256, 0, stream>>>(h, batch, pool, cnt);
    cg_head_kernel<<<1, 128, 0, stream>>>(pool, cnt, W2, b2, W3, b3, out);
}
